// Decoder_43696997269791
// MI455X (gfx1250) — matrix operand order repaired and checked by reading, not compiled, not run on hardware
//
#include <hip/hip_runtime.h>
#include <hip/hip_bf16.h>

typedef __attribute__((ext_vector_type(16))) __bf16 v16bf;
typedef __attribute__((ext_vector_type(8)))  float  v8f;

#define B_BATCH 64
#define L_SEQ   700
#define D_DIM   512
#define L_PAD   704   // 44 tiles of 16; bf16 buffer padded to this, pad rows = 0

__device__ __forceinline__ unsigned short f2bf_rne(float x) {
    unsigned u = __float_as_uint(x);
    unsigned r = (u + 0x7FFFu + ((u >> 16) & 1u)) >> 16;
    return (unsigned short)r;
}

// fast tanh: 1 - 2/(exp(2x)+1); exact saturation at +/-1, hardware trans ops only
__device__ __forceinline__ float fast_tanh(float x) {
    float e = __builtin_amdgcn_exp2f(x * 2.885390081777927f); // 2*log2(e)
    float r = __builtin_amdgcn_rcpf(e + 1.0f);
    return __builtin_fmaf(-2.0f, r, 1.0f);
}

// ------------- Kernel 1: fp32 -> bf16, padded to 704 rows/batch (pad rows = 0) ----
__global__ __launch_bounds__(256) void convert_bf16_kernel(
    const float* __restrict__ E, unsigned short* __restrict__ Ebf, int total4) {
    int i = blockIdx.x * blockDim.x + threadIdx.x;
    if (i >= total4) return;
    const int per_batch = (L_PAD * D_DIM) / 4;       // 90112
    const int b   = i / per_batch;
    const int rem = i - b * per_batch;
    const int r   = rem >> 7;                        // / (512/4)
    const int d4  = rem & 127;
    ushort4 o;
    if (r < L_SEQ) {
        float4 f = ((const float4*)(E + ((size_t)b * L_SEQ + r) * D_DIM))[d4];
        o.x = f2bf_rne(f.x); o.y = f2bf_rne(f.y);
        o.z = f2bf_rne(f.z); o.w = f2bf_rne(f.w);
    } else {
        o.x = o.y = o.z = o.w = 0;
    }
    ((ushort4*)(Ebf + ((size_t)b * L_PAD + r) * D_DIM))[d4] = o;
}

// ------------- Kernel 2: S = E·Eᵀ/√512, tanh, *mask, column-sum ------------------
// grid: (11 strips of 64 cols, 64 batches), block: 256 threads = 8 waves.
// wave w: mt = w&3 (16-col tile of strip), lhalf = w>>2 (16-row half of 32-row step).
// B tile (rows m0..m0+15 of padded E, all K=512) hoisted into 128 VGPRs.
__global__ __launch_bounds__(256) void attn_colsum_kernel(
    const unsigned short* __restrict__ Ebf,
    const float* __restrict__ mask,
    float* __restrict__ aOut) {
    const int b       = blockIdx.y;
    const int m0strip = blockIdx.x * 64;
    const int tid     = threadIdx.x;
    const int wave    = tid >> 5;
    const int lane    = tid & 31;
    const int mt      = wave & 3;
    const int lhalf   = wave >> 2;
    const int laneLo  = lane & 15;
    const int laneHi  = lane >> 4;
    const int m0      = m0strip + 16 * mt;

    __shared__ float colsum[64];
    if (tid < 64) colsum[tid] = 0.f;
    __syncthreads();

    // ---- preload B: B[k,n] = E[m0+n, k]; lane n<16 holds K 0..15 of each chunk,
    //      lane n+16 holds K 16..31 (contiguous 32B per lane). Unconditional: padded.
    v16bf Breg[16];
    {
        const uint4* rowq =
            (const uint4*)(Ebf + ((size_t)b * L_PAD + (m0 + laneLo)) * D_DIM);
#pragma unroll
        for (int kk = 0; kk < 16; ++kk) {
            union { v16bf v; uint4 q[2]; } u;
            u.q[0] = rowq[kk * 4 + 2 * laneHi];
            u.q[1] = rowq[kk * 4 + 2 * laneHi + 1];
            Breg[kk] = u.v;
        }
    }

    const float invT = 0.044194173824159216f; // 1/sqrt(512)
    float colacc = 0.f;

    for (int l0 = 0; l0 < L_SEQ; l0 += 32) {
        // A tile rows l0+16*lhalf .. +15 (pad rows are zeros -> no guard needed)
        const uint4* rowq =
            (const uint4*)(Ebf + ((size_t)b * L_PAD + (l0 + 16 * lhalf + laneLo)) * D_DIM);

        v8f c0 = {}, c1 = {};
#pragma unroll
        for (int kk = 0; kk < 16; kk += 2) {
            union { v16bf v; uint4 q[2]; } ua, ub;
            ua.q[0] = rowq[kk * 4 + 2 * laneHi];       // K = kk*32 + 16*laneHi .. +7
            ua.q[1] = rowq[kk * 4 + 2 * laneHi + 1];   // K = kk*32 + 16*laneHi + 8 .. +15
            ub.q[0] = rowq[(kk + 1) * 4 + 2 * laneHi];
            ub.q[1] = rowq[(kk + 1) * 4 + 2 * laneHi + 1];
            c0 = __builtin_amdgcn_wmma_f32_16x16x32_bf16(
                     false, ua.v, false, Breg[kk], (short)0, c0, false, false);
            c1 = __builtin_amdgcn_wmma_f32_16x16x32_bf16(
                     false, ub.v, false, Breg[kk + 1], (short)0, c1, false, false);
        }

        // ---- epilogue: tanh(c/√512)*mask, reduce the 8 rows this lane owns ----
        const int rowBase = l0 + 16 * lhalf + 8 * laneHi;
        const int col     = m0 + laneLo;
        const float* mptr = mask + ((size_t)b * L_SEQ + rowBase) * (size_t)L_SEQ + col;
        float s = 0.f;
#pragma unroll
        for (int j = 0; j < 8; ++j) {
            const int   row = rowBase + j;
            const float mv  = (row < L_SEQ && col < L_SEQ) ? mptr[(size_t)j * L_SEQ] : 0.f;
            s += fast_tanh((c0[j] + c1[j]) * invT) * mv;
        }
        colacc += s;
    }

    // combine the two 8-row lane halves (same column: lane ^ 16)
    colacc += __shfl_xor(colacc, 16);
    if (laneHi == 0) atomicAdd(&colsum[mt * 16 + laneLo], colacc);
    __syncthreads();
    if (tid < 64) aOut[(size_t)b * L_PAD + m0strip + tid] = colsum[tid];
}

// ------------- Kernel 3: seq1 matvec + separable conv + final tanh ---------------
__global__ __launch_bounds__(512) void finalize_kernel(
    const float* __restrict__ user,
    const float* __restrict__ E,
    const float* __restrict__ aArr,
    const float* __restrict__ conv_w,   // (3,3)
    const float* __restrict__ conv_b,   // (1,)
    const float* __restrict__ conv3_w,  // (700,)
    const float* __restrict__ conv3_b,  // (1,)
    float* __restrict__ out) {
    const int b = blockIdx.x, tid = threadIdx.x;

    __shared__ float a_s[L_PAD];
    __shared__ float wpad[L_PAD];           // wpad[i+1] = w3[i], zero-padded ends
    __shared__ float tM[D_DIM], t0s[D_DIM], tP[D_DIM];
    __shared__ float sW3;

    for (int i = tid; i < L_PAD; i += 512) {
        a_s[i]  = (i < L_SEQ) ? aArr[(size_t)b * L_PAD + i] : 0.f;
        wpad[i] = (i >= 1 && i <= L_SEQ) ? conv3_w[i - 1] : 0.f;
    }
    if (tid == 0) {
        float s = 0.f;
        for (int h = 0; h < L_SEQ; ++h) s += conv3_w[h];
        sW3 = s;
    }
    __syncthreads();

    const int d = tid;
    float acc1 = 0.f, am = 0.f, a0 = 0.f, ap = 0.f;
    const float* Eb = E + (size_t)b * L_SEQ * D_DIM + d;
    for (int m = 0; m < L_SEQ; ++m) {
        const float e = Eb[(size_t)m * D_DIM];   // coalesced across d
        acc1 += a_s[m]      * e;   // seq1 matvec
        am   += wpad[m + 2] * e;   // t_{-1}: coeff w3[m+1]
        a0   += wpad[m + 1] * e;   // t_{0} : coeff w3[m]
        ap   += wpad[m]     * e;   // t_{+1}: coeff w3[m-1]
    }
    tM[d] = am; t0s[d] = a0; tP[d] = ap;
    __syncthreads();

    float cw[3][3];
#pragma unroll
    for (int i = 0; i < 9; ++i) cw[i / 3][i % 3] = conv_w[i];

    float s2 = conv_b[0] * sW3 + conv3_b[0];
#pragma unroll
    for (int kw = 0; kw < 3; ++kw) {
        const int dd = d + kw - 1;
        if (dd >= 0 && dd < D_DIM)
            s2 += cw[0][kw] * tM[dd] + cw[1][kw] * t0s[dd] + cw[2][kw] * tP[dd];
    }

    out[(size_t)b * D_DIM + d] =
        tanhf(user[(size_t)b * D_DIM + d] + acc1 * (1.f / 1400.f) + 2.f * s2);
}

extern "C" void kernel_launch(void* const* d_in, const int* in_sizes, int n_in,
                              void* d_out, int out_size, void* d_ws, size_t ws_size,
                              hipStream_t stream) {
    const float* user    = (const float*)d_in[0];
    /* d_in[1] = embeddings: unused by the reference */
    const float* enc     = (const float*)d_in[2];
    const float* mask    = (const float*)d_in[3];
    const float* conv_w  = (const float*)d_in[4];
    const float* conv_b  = (const float*)d_in[5];
    const float* conv3_w = (const float*)d_in[6];
    const float* conv3_b = (const float*)d_in[7];
    float* out = (float*)d_out;

    unsigned short* Ebf  = (unsigned short*)d_ws;
    float*          aArr = (float*)((char*)d_ws + (size_t)B_BATCH * L_PAD * D_DIM * 2);

    const int total4 = (B_BATCH * L_PAD * D_DIM) / 4;
    convert_bf16_kernel<<<(total4 + 255) / 256, 256, 0, stream>>>(enc, Ebf, total4);
    attn_colsum_kernel<<<dim3((L_SEQ + 63) / 64, B_BATCH), 256, 0, stream>>>(Ebf, mask, aArr);
    finalize_kernel<<<B_BATCH, 512, 0, stream>>>(user, enc, aArr, conv_w, conv_b,
                                                 conv3_w, conv3_b, out);
}
